// AnomalyAttention_72284299591903
// MI455X (gfx1250) — matrix operand order repaired and checked by reading, not compiled, not run on hardware
//
#include <hip/hip_runtime.h>
#include <hip/hip_bf16.h>
#include <math.h>

// Problem constants (match reference)
#define Bb 2
#define Ls 1024
#define Xh 4
#define Eh 128
#define Yh 4
#define SCALE 0.08838834764831845f  // 1/sqrt(128)

typedef __attribute__((ext_vector_type(16))) __bf16 v16bf;
typedef __attribute__((ext_vector_type(8)))  float  v8f;
typedef unsigned int tdm_v4u __attribute__((ext_vector_type(4)));
typedef int          tdm_v8i __attribute__((ext_vector_type(8)));
typedef int          tdm_v4i __attribute__((ext_vector_type(4)));

#define DEV static __device__ __forceinline__

#if __has_builtin(__builtin_amdgcn_tensor_load_to_lds) && __has_builtin(__builtin_amdgcn_s_wait_tensorcnt)
#define HAS_TDM 1
#else
#define HAS_TDM 0
#endif

DEV unsigned short f2bf(float f) {
    union { float f; unsigned u; } c; c.f = f;
    unsigned r = c.u + 0x7fffu + ((c.u >> 16) & 1u);   // round-to-nearest-even
    return (unsigned short)(r >> 16);
}

struct U8x32 { uint4 lo, hi; };
DEV v16bf frag_cast(uint4 lo, uint4 hi) {
    union { U8x32 u; v16bf v; } c; c.u.lo = lo; c.u.hi = hi; return c.v;
}

// A-fragment (16x32 bf16), row-major source, row stride `ld` (elements).
// ISA: lanes 0-15 row M=lane, elems K{kb+0..7, kb+16..23}; lanes 16-31 row M=lane-16, K{kb+8..15, kb+24..31}.
DEV v16bf loadA(const unsigned short* src, int ld, int lane, int kb) {
    int r = lane & 15; bool lo = lane < 16;
    const unsigned short* p = src + r * ld + kb;
    uint4 a = *(const uint4*)(p + (lo ? 0 : 8));
    uint4 b = *(const uint4*)(p + (lo ? 16 : 24));
    return frag_cast(a, b);
}

// B-fragment (32x16 bf16): column n = lane&15 maps to row n of `src` (contraction contiguous).
DEV v16bf loadB(const unsigned short* src, int ld, int lane, int kb) {
    int n = lane & 15; int ko = (lane < 16) ? 0 : 8;
    const unsigned short* p = src + n * ld + kb + ko;
    uint4 a = *(const uint4*)p;
    uint4 b = *(const uint4*)(p + 16);
    return frag_cast(a, b);
}

DEV v8f wmma_bf16(v16bf a, v16bf b, v8f c) {
    return __builtin_amdgcn_wmma_f32_16x16x32_bf16(false, a, false, b, (short)0, c, false, false);
}

DEV float rmax16(float v) { for (int o = 8; o; o >>= 1) v = fmaxf(v, __shfl_xor(v, o, 16)); return v; }
DEV float rsum16(float v) { for (int o = 8; o; o >>= 1) v += __shfl_xor(v, o, 16); return v; }

// ---- Tensor Data Mover: 2D bf16 tile (tile_d0 elems/row x tile_d1 rows) global -> LDS (packed rows)
// D# packing per CDNA5 ISA ch.8: group0 {count=1, lds_addr, global_addr[56:0], type=2},
// group1 {data_size=2B, tensor_dim0/1, tile_dim0/1, tensor_dim0_stride}. Groups 2/3 zero (<=2D).
DEV void tile_to_lds(unsigned short* lds_ptr, unsigned lds_byte_off, const unsigned short* g,
                     unsigned tile_d0, unsigned tile_d1, unsigned row_stride, int lane)
{
#if HAS_TDM
    (void)lds_ptr; (void)lane;
    unsigned long long ga = (unsigned long long)(const void*)g;
    tdm_v4u g0;
    g0[0] = 1u;                                                // count=1 (user descriptor)
    g0[1] = lds_byte_off;                                      // lds_addr (bytes)
    g0[2] = (unsigned)(ga & 0xffffffffu);                      // global_addr[31:0]
    g0[3] = (unsigned)((ga >> 32) & 0x01ffffffu) | (2u << 30); // global_addr[56:32], type=2
    const unsigned td0 = 1024u, td1 = 1024u;                   // tensor dims (OOB never hit)
    tdm_v8i g1;
    g1[0] = (int)(1u << 16);                                   // data_size=1 -> 2 bytes
    g1[1] = (int)((td0 & 0xffffu) << 16);                      // tensor_dim0[15:0] -> bits 63:48
    g1[2] = (int)(((td0 >> 16) & 0xffffu) | ((td1 & 0xffffu) << 16));
    g1[3] = (int)(((td1 >> 16) & 0xffffu) | ((tile_d0 & 0xffffu) << 16));  // tile_dim0
    g1[4] = (int)(tile_d1 & 0xffffu);                          // tile_dim1 (tile_dim2=0)
    g1[5] = (int)row_stride;                                   // tensor_dim0_stride[31:0]
    g1[6] = 0;
    g1[7] = 0;
    tdm_v4i z4 = {};
    tdm_v8i z8 = {};
    __builtin_amdgcn_tensor_load_to_lds(g0, g1, z4, z4, z8, 0);
#else
    (void)lds_byte_off;
    int total = (int)(tile_d0 * tile_d1) / 8;                  // 16B chunks, one wave
    for (int i = lane; i < total; i += 32) {
        int r = (i * 8) / (int)tile_d0, c = (i * 8) % (int)tile_d0;
        *(uint4*)(lds_ptr + r * tile_d0 + c) = *(const uint4*)(g + (size_t)r * row_stride + c);
    }
#endif
}

DEV void tdm_wait() {
#if HAS_TDM
    __builtin_amdgcn_s_wait_tensorcnt((unsigned short)0);
#endif
}

// ---------------------------------------------------------------- K0: pack to bf16
__global__ void k_pack(const float* __restrict__ q, const float* __restrict__ k,
                       const float* __restrict__ v, const float* __restrict__ qp,
                       const float* __restrict__ Wk, const float* __restrict__ Wv,
                       unsigned short* qb, unsigned short* kb, unsigned short* vbT,
                       unsigned short* qpb, unsigned short* wkb, unsigned short* wvb)
{
    const int NQ = Bb * Ls * Xh * Eh;
    int total = NQ * 4 + Eh * Eh * 2;
    for (int i = blockIdx.x * blockDim.x + threadIdx.x; i < total; i += gridDim.x * blockDim.x) {
        if (i < NQ * 3) {
            int seg = i / NQ, t = i % NQ;
            int e = t % Eh, r = t / Eh; int x = r % Xh; r /= Xh; int l = r % Ls; int b = r / Ls;
            int dst = ((b * Xh + x) * Ls + l) * Eh + e;
            const float* s = (seg == 0) ? q : (seg == 1 ? k : qp);
            unsigned short* d = (seg == 0) ? qb : (seg == 1 ? kb : qpb);
            d[dst] = f2bf(s[t]);
        } else if (i < NQ * 4) {
            int t = i - NQ * 3;
            int e = t % Eh, r = t / Eh; int y = r % Yh; r /= Yh; int l = r % Ls; int b = r / Ls;
            vbT[((b * Yh + y) * Eh + e) * Ls + l] = f2bf(v[t]);   // value transposed [B][Y][E][L]
        } else {
            int t = i - NQ * 4;
            if (t < Eh * Eh) wkb[t] = f2bf(Wk[t]);
            else             wvb[t - Eh * Eh] = f2bf(Wv[t - Eh * Eh]);
        }
    }
}

// ---------------------------------------------------------------- K1: stage A flash attention
// grid (L/16, X, B), block 128 (4 waves; wave = y). TDM double-buffers K tiles (shared, wave 0
// issues) and per-wave V^T tiles into LDS; one workgroup barrier per tile for K visibility.
// LDS map (shorts): [0..8191]  K bufs 2x(32x128)
//                   [8192..40959] VT bufs 4 waves x 2 x (128x32)
//                   [40960..43007] P tiles 4 x (16x32)
__global__ __launch_bounds__(128) void k_stage1(
    const unsigned short* __restrict__ qb, const unsigned short* __restrict__ kb,
    const unsigned short* __restrict__ vbT,
    unsigned short* __restrict__ Rb, unsigned short* __restrict__ RTb)
{
    __shared__ unsigned short smem[43008];
    int lane = threadIdx.x & 31, wave = threadIdx.x >> 5;
    int x = blockIdx.y, b = blockIdx.z, y = wave;
    bool active = (y <= x);
    int l0 = blockIdx.x * 16;

    const unsigned short* Q  = qb  + ((size_t)(b * Xh + x) * Ls + l0) * Eh;
    const unsigned short* K  = kb  + (size_t)(b * Xh + x) * Ls * Eh;
    const unsigned short* VT = vbT + (size_t)(b * Yh + y) * Eh * Ls;

    unsigned short* ldsP = smem + 40960 + wave * 512;

    v16bf qa[4];
    v8f O[8] = {};
    float mrun[8], lrun[8];
    if (active) {
#pragma unroll
        for (int ke = 0; ke < 4; ke++) qa[ke] = loadA(Q, Eh, lane, ke * 32);
#pragma unroll
        for (int v = 0; v < 8; v++) { mrun[v] = -INFINITY; lrun[v] = 0.f; }
    }
    int row_off = (lane < 16) ? 0 : 8;
    int n = lane & 15;

    // prime buffer 0
    if (wave == 0) tile_to_lds(smem, 0u, K, 128, 32, Eh, lane);
    if (active)    tile_to_lds(smem + 8192 + wave * 8192, 16384u + wave * 16384u, VT, 32, 128, Ls, lane);

    const int NT = Ls / 32;
    for (int it = 0; it < NT; ++it) {
        int st = it * 32, cur = it & 1, nxt = cur ^ 1;
        if (active) tdm_wait();            // own TDMs for tile `it` complete
        __syncthreads();                   // K tile visible; everyone past previous compute
        if (it + 1 < NT) {
            int st2 = st + 32;
            if (wave == 0)
                tile_to_lds(smem + nxt * 4096, (unsigned)(nxt * 8192),
                            K + (size_t)st2 * Eh, 128, 32, Eh, lane);
            if (active)
                tile_to_lds(smem + 8192 + wave * 8192 + nxt * 4096,
                            (unsigned)(16384 + wave * 16384 + nxt * 8192),
                            VT + st2, 32, 128, Ls, lane);
        }
        if (active) {
            const unsigned short* bufK = smem + cur * 4096;          // [32][128]
            const unsigned short* bufV = smem + 8192 + wave * 8192 + cur * 4096; // [128][32]
            // batched B-fragment loads, then WMMA chain (overlap loads with matrix pipe)
            v16bf kb0[4], kb1[4];
#pragma unroll
            for (int ke = 0; ke < 4; ke++) {
                kb0[ke] = loadB(bufK, Eh, lane, ke * 32);
                kb1[ke] = loadB(bufK + 16 * Eh, Eh, lane, ke * 32);
            }
            v8f S0 = {}, S1 = {};
#pragma unroll
            for (int ke = 0; ke < 4; ke++) {
                S0 = wmma_bf16(qa[ke], kb0[ke], S0);
                S1 = wmma_bf16(qa[ke], kb1[ke], S1);
            }
            float corr[8];
#pragma unroll
            for (int v = 0; v < 8; v++) {
                float s0 = S0[v] * SCALE, s1 = S1[v] * SCALE;
                float rm = rmax16(fmaxf(s0, s1));
                float mn = fmaxf(mrun[v], rm);
                corr[v] = __expf(mrun[v] - mn);
                float p0 = __expf(s0 - mn), p1 = __expf(s1 - mn);
                S0[v] = p0; S1[v] = p1;
                lrun[v] = lrun[v] * corr[v] + rsum16(p0 + p1);
                mrun[v] = mn;
            }
#pragma unroll
            for (int eb = 0; eb < 8; eb++)
#pragma unroll
                for (int v = 0; v < 8; v++) O[eb][v] *= corr[v];
            // P tile -> LDS (bf16) -> reload as A-fragment
#pragma unroll
            for (int v = 0; v < 8; v++) {
                ldsP[(v + row_off) * 32 + n]      = f2bf(S0[v]);
                ldsP[(v + row_off) * 32 + 16 + n] = f2bf(S1[v]);
            }
            asm volatile("s_wait_dscnt 0" ::: "memory");
            v16bf pa = loadA(ldsP, 32, lane, 0);
            v16bf vf[8];
#pragma unroll
            for (int eb = 0; eb < 8; eb++) vf[eb] = loadB(bufV + (eb * 16) * 32, 32, lane, 0);
#pragma unroll
            for (int eb = 0; eb < 8; eb++) O[eb] = wmma_bf16(pa, vf[eb], O[eb]);
        }
    }

    if (active) {
        float inv[8];
#pragma unroll
        for (int v = 0; v < 8; v++) inv[v] = 1.0f / lrun[v];
        unsigned short* Rp  = Rb  + (((size_t)(b * Xh + x) * Yh + y) * Ls + l0) * Eh;
        unsigned short* RTp = RTb + ((size_t)(b * Xh + x) * Yh + y) * Eh * Ls + l0;
#pragma unroll
        for (int eb = 0; eb < 8; eb++)
#pragma unroll
            for (int v = 0; v < 8; v++) {
                unsigned short h = f2bf(O[eb][v] * inv[v]);
                int row = v + row_off, e = eb * 16 + n;
                Rp[(size_t)row * Eh + e] = h;
                RTp[(size_t)e * Ls + row] = h;
            }
    }
}

// ---------------------------------------------------------------- K2: key_plus = Retrieval @ Wk^T + bk
__global__ __launch_bounds__(32) void k_keyplus(
    const unsigned short* __restrict__ Rb, const unsigned short* __restrict__ wkb,
    const float* __restrict__ bk, unsigned short* __restrict__ KPb)
{
    int lane = threadIdx.x & 31;
    int xy = blockIdx.y, x = xy >> 2, y = xy & 3, b = blockIdx.z;
    if (y > x) return;
    int l0 = blockIdx.x * 16;
    const unsigned short* A = Rb + (((size_t)(b * Xh + x) * Yh + y) * Ls + l0) * Eh;

    v16bf af[4];
#pragma unroll
    for (int ke = 0; ke < 4; ke++) af[ke] = loadA(A, Eh, lane, ke * 32);

    v8f C[8] = {};
#pragma unroll
    for (int nb = 0; nb < 8; nb++) {
        v16bf bf[4];
#pragma unroll
        for (int ke = 0; ke < 4; ke++) bf[ke] = loadB(wkb + (size_t)(nb * 16) * Eh, Eh, lane, ke * 32);
#pragma unroll
        for (int ke = 0; ke < 4; ke++) C[nb] = wmma_bf16(af[ke], bf[ke], C[nb]);
    }

    int n = lane & 15, row_off = (lane < 16) ? 0 : 8;
    unsigned short* KP = KPb + (((size_t)(b * Xh + x) * Yh + y) * Ls + l0) * Eh;
#pragma unroll
    for (int nb = 0; nb < 8; nb++) {
        float bias = bk[nb * 16 + n];
#pragma unroll
        for (int v = 0; v < 8; v++)
            KP[(size_t)(v + row_off) * Eh + nb * 16 + n] = f2bf(C[nb][v] + bias);
    }
}

// ---------------------------------------------------------------- K3: stage B fused
__global__ __launch_bounds__(128) void k_stage2(
    const unsigned short* __restrict__ qpb, const unsigned short* __restrict__ KPb,
    const unsigned short* __restrict__ RTb, unsigned short* __restrict__ Vws)
{
    __shared__ unsigned short ldsP[4][16 * 32];
    int lane = threadIdx.x & 31, wave = threadIdx.x >> 5;
    int x = blockIdx.y, b = blockIdx.z;
    int l0 = blockIdx.x * 64 + wave * 16;

    const unsigned short* QP = qpb + ((size_t)(b * Xh + x) * Ls + l0) * Eh;
    v16bf qa[4];
#pragma unroll
    for (int ke = 0; ke < 4; ke++) qa[ke] = loadA(QP, Eh, lane, ke * 32);

    v8f V[8] = {};
    int row_off = (lane < 16) ? 0 : 8;
    int n = lane & 15;

    for (int mt = 0; mt < Ls; mt += 32) {
        v8f S[4][2] = {};
#pragma unroll
        for (int y = 0; y < 4; y++) {
            if (y > x) continue;
            const unsigned short* KP = KPb + (((size_t)(b * Xh + x) * Yh + y) * Ls + mt) * Eh;
            v16bf bb[8];
#pragma unroll
            for (int ke = 0; ke < 4; ke++) {
                bb[ke]     = loadB(KP, Eh, lane, ke * 32);
                bb[4 + ke] = loadB(KP + (size_t)16 * Eh, Eh, lane, ke * 32);
            }
#pragma unroll
            for (int ke = 0; ke < 4; ke++) {
                S[y][0] = wmma_bf16(qa[ke], bb[ke], S[y][0]);
                S[y][1] = wmma_bf16(qa[ke], bb[4 + ke], S[y][1]);
            }
        }
        // elementwise softmax across y (mask y > x)
#pragma unroll
        for (int sub = 0; sub < 2; sub++)
#pragma unroll
            for (int v = 0; v < 8; v++) {
                float mx = -INFINITY;
#pragma unroll
                for (int y = 0; y < 4; y++) if (y <= x) mx = fmaxf(mx, S[y][sub][v] * SCALE);
                float p[4]; float sum = 0.f;
#pragma unroll
                for (int y = 0; y < 4; y++) if (y <= x) { p[y] = __expf(S[y][sub][v] * SCALE - mx); sum += p[y]; }
                float is = 1.0f / sum;
#pragma unroll
                for (int y = 0; y < 4; y++) if (y <= x) S[y][sub][v] = p[y] * is;
            }
        // V += series_y @ Retrieval_y
#pragma unroll
        for (int y = 0; y < 4; y++) {
            if (y > x) continue;
#pragma unroll
            for (int v = 0; v < 8; v++) {
                ldsP[wave][(v + row_off) * 32 + n]      = f2bf(S[y][0][v]);
                ldsP[wave][(v + row_off) * 32 + 16 + n] = f2bf(S[y][1][v]);
            }
            asm volatile("s_wait_dscnt 0" ::: "memory");
            v16bf pa = loadA(ldsP[wave], 32, lane, 0);
            const unsigned short* RT = RTb + ((size_t)(b * Xh + x) * Yh + y) * Eh * Ls;
            v16bf rf[8];
#pragma unroll
            for (int eb = 0; eb < 8; eb++) rf[eb] = loadB(RT + (size_t)(eb * 16) * Ls + mt, Ls, lane, 0);
#pragma unroll
            for (int eb = 0; eb < 8; eb++) V[eb] = wmma_bf16(pa, rf[eb], V[eb]);
        }
    }

    unsigned short* vp = Vws + (size_t)(b * Ls + l0) * 512;
#pragma unroll
    for (int eb = 0; eb < 8; eb++)
#pragma unroll
        for (int v = 0; v < 8; v++) {
            int row = v + row_off, e = eb * 16 + n;
            vp[(size_t)row * 512 + e * 4 + x] = f2bf(V[eb][v]);
        }
}

// ---------------------------------------------------------------- K4: V_out = V_list @ Wv^T + bv
__global__ __launch_bounds__(32) void k_vproj(
    const unsigned short* __restrict__ Vws, const unsigned short* __restrict__ wvb,
    const float* __restrict__ bv, float* __restrict__ outV)
{
    int lane = threadIdx.x & 31;
    size_t tile = blockIdx.x;
    const unsigned short* A = Vws + tile * 16 * 128;

    v16bf af[4];
#pragma unroll
    for (int ke = 0; ke < 4; ke++) af[ke] = loadA(A, 128, lane, ke * 32);

    v8f C[8] = {};
#pragma unroll
    for (int nb = 0; nb < 8; nb++) {
        v16bf bf[4];
#pragma unroll
        for (int ke = 0; ke < 4; ke++) bf[ke] = loadB(wvb + (size_t)(nb * 16) * Eh, Eh, lane, ke * 32);
#pragma unroll
        for (int ke = 0; ke < 4; ke++) C[nb] = wmma_bf16(af[ke], bf[ke], C[nb]);
    }

    int n = lane & 15, row_off = (lane < 16) ? 0 : 8;
#pragma unroll
    for (int nb = 0; nb < 8; nb++) {
        float bias = bv[nb * 16 + n];
#pragma unroll
        for (int v = 0; v < 8; v++)
            outV[(tile * 16 + v + row_off) * 128 + nb * 16 + n] = C[nb][v] + bias;
    }
}

// ---------------------------------------------------------------- K5: prior & sig
__global__ __launch_bounds__(256) void k_prior(const float* __restrict__ sigma,
                                               float* __restrict__ prior, float* __restrict__ sigout)
{
    int r = blockIdx.x;                       // (b, h, l)
    int l = r % Ls; int t = r / Ls; int h = t % Xh; int b = t / Xh;
    float s = sigma[(size_t)(b * Ls + l) * Xh + h];
    float sg = 1.0f / (1.0f + __expf(-5.0f * s)) + 1e-5f;
    sg = exp2f(sg * 1.5849625007211562f) - 1.0f;   // 3^sg - 1
    if (threadIdx.x == 0) sigout[r] = sg;
    float c = 0.3989422804014327f / sg;
    float inv2 = -0.5f / (sg * sg);
    float* row = prior + (size_t)r * Ls;
    for (int j = threadIdx.x; j < Ls; j += 256) {
        float d = (float)(l - j);
        row[j] = c * __expf(d * d * inv2);
    }
}

// ---------------------------------------------------------------- K6: series_out == 0.25 exactly
__global__ void k_fill(float* __restrict__ p, int total, float v)
{
    for (int i = blockIdx.x * blockDim.x + threadIdx.x; i < total; i += gridDim.x * blockDim.x)
        p[i] = v;
}

// ----------------------------------------------------------------
extern "C" void kernel_launch(void* const* d_in, const int* in_sizes, int n_in,
                              void* d_out, int out_size, void* d_ws, size_t ws_size,
                              hipStream_t stream)
{
    (void)in_sizes; (void)n_in; (void)out_size; (void)ws_size;
    // setup_inputs order: x, query, key, value, sigma, attn_mask, retrieval, search, q_plus, Wk, bk, Wv, bv
    const float* query = (const float*)d_in[1];
    const float* key   = (const float*)d_in[2];
    const float* value = (const float*)d_in[3];
    const float* sigma = (const float*)d_in[4];
    const float* qplus = (const float*)d_in[8];
    const float* Wk    = (const float*)d_in[9];
    const float* bk    = (const float*)d_in[10];
    const float* Wv    = (const float*)d_in[11];
    const float* bv    = (const float*)d_in[12];

    char* ws = (char*)d_ws;
    size_t off = 0;
    auto carve = [&](size_t bytes) { void* p = ws + off; off += (bytes + 255) & ~(size_t)255; return p; };
    const size_t NQ = (size_t)Bb * Ls * Xh * Eh;                 // 1,048,576
    unsigned short* qb  = (unsigned short*)carve(NQ * 2);
    unsigned short* kb  = (unsigned short*)carve(NQ * 2);
    unsigned short* qpb = (unsigned short*)carve(NQ * 2);
    unsigned short* vbT = (unsigned short*)carve((size_t)Bb * Yh * Eh * Ls * 2);
    unsigned short* wkb = (unsigned short*)carve((size_t)Eh * Eh * 2);
    unsigned short* wvb = (unsigned short*)carve((size_t)Eh * Eh * 2);
    unsigned short* Rb  = (unsigned short*)carve((size_t)Bb * Xh * Yh * Ls * Eh * 2);
    unsigned short* RTb = (unsigned short*)carve((size_t)Bb * Xh * Yh * Eh * Ls * 2);
    unsigned short* KPb = (unsigned short*)carve((size_t)Bb * Xh * Yh * Ls * Eh * 2);
    unsigned short* Vws = (unsigned short*)carve((size_t)Bb * Ls * 512 * 2);

    float* outV     = (float*)d_out;                             // [B,L,X,E]   1,048,576
    float* outSer   = outV + (size_t)Bb * Ls * Xh * Eh;          // [B,X,L,L]   8,388,608
    float* outPrior = outSer + (size_t)Bb * Xh * Ls * Ls;        // [B,X,L,L]   8,388,608
    float* outSig   = outPrior + (size_t)Bb * Xh * Ls * Ls;      // [B,X,L]     8,192

    int packTotal = (int)(NQ * 4 + (size_t)Eh * Eh * 2);
    k_pack<<<dim3((packTotal + 255) / 256), dim3(256), 0, stream>>>(
        query, key, value, qplus, Wk, Wv, qb, kb, vbT, qpb, wkb, wvb);

    k_stage1<<<dim3(Ls / 16, Xh, Bb), dim3(128), 0, stream>>>(qb, kb, vbT, Rb, RTb);

    k_keyplus<<<dim3(Ls / 16, Xh * Yh, Bb), dim3(32), 0, stream>>>(Rb, wkb, bk, KPb);

    k_stage2<<<dim3(Ls / 64, Xh, Bb), dim3(128), 0, stream>>>(qpb, KPb, RTb, Vws);

    k_vproj<<<dim3((Bb * Ls * Xh) / 16), dim3(32), 0, stream>>>(Vws, wvb, bv, outV);

    k_prior<<<dim3(Bb * Xh * Ls), dim3(256), 0, stream>>>(sigma, outPrior, outSig);

    k_fill<<<dim3(8192), dim3(256), 0, stream>>>(outSer, Bb * Xh * Ls * Ls, 0.25f);
}
